// QKVAttention_29738353558227
// MI455X (gfx1250) — compile-verified, matrix operand order checked by reading, not run on hardware
//
#include <hip/hip_runtime.h>

// QKV attention, flash-style, f16 WMMA with f32 accumulate. gfx1250 wave32.
//   qkv: [64, 192, 2048] fp32  (q=ch 0..63, k=64..127, v=128..191; token dim contiguous)
//   out: [64,  64, 2048] fp32
// Per block: one head (b) x 64-query tile. 4 waves, each owns 16 query rows.
// Register-pipelined double-buffered 64-key chunks:
//   18 wmma / chunk (8 QK + 2 row-sum + 8 PV), softmax max-reduce on DPP16 (no DS).

typedef __attribute__((ext_vector_type(16))) _Float16 v16h;
typedef __attribute__((ext_vector_type(8)))  _Float16 v8h;
typedef __attribute__((ext_vector_type(8)))  float    v8f;
typedef __attribute__((ext_vector_type(4)))  float    v4f;

#define NB      64
#define C3CH    192
#define CH      64
#define TT      2048
#define BR      64      // query rows per block
#define BC      64      // keys per chunk
#define QPITCH  72      // halves; 144 B rows (16B aligned)
#define KPITCH  72
#define VPITCH  72
#define PPITCH  72
#define OPITCH  68      // floats; 272 B rows

// Build a 16x32 f16 A/B fragment from an LDS row, per CDNA5 16-bit layout:
//   lanes 0-15 : K = kbase+0..7  (v0-3),  kbase+16..23 (v4-7)
//   lanes 16-31: K = kbase+8..15 (v0-3),  kbase+24..31 (v4-7)
__device__ __forceinline__ v16h load_frag16(const _Float16* row, int kbase, int lh) {
  v8h lo = *(const v8h*)(row + kbase + lh * 8);
  v8h hi = *(const v8h*)(row + kbase + 16 + lh * 8);
  v16h a;
#pragma unroll
  for (int i = 0; i < 8; ++i) { a[i] = lo[i]; a[i + 8] = hi[i]; }
  return a;
}

// Max-reduce across each 16-lane row using DPP16 permutes (pure VALU, no DS):
//   xor1 = quad_perm(1,0,3,2)=0xB1, xor2 = quad_perm(2,3,0,1)=0x4E,
//   quads paired by row_half_mirror=0x141, halves paired by row_mirror=0x140.
__device__ __forceinline__ float rowmax16(float x) {
  int t;
  t = __builtin_amdgcn_update_dpp(0, __float_as_int(x), 0xB1,  0xF, 0xF, true);
  x = fmaxf(x, __int_as_float(t));
  t = __builtin_amdgcn_update_dpp(0, __float_as_int(x), 0x4E,  0xF, 0xF, true);
  x = fmaxf(x, __int_as_float(t));
  t = __builtin_amdgcn_update_dpp(0, __float_as_int(x), 0x141, 0xF, 0xF, true);
  x = fmaxf(x, __int_as_float(t));
  t = __builtin_amdgcn_update_dpp(0, __float_as_int(x), 0x140, 0xF, 0xF, true);
  x = fmaxf(x, __int_as_float(t));
  return x;
}

__device__ __forceinline__ void load_chunk(const float* __restrict__ src, v4f* reg) {
#pragma unroll
  for (int i = 0; i < 8; ++i) reg[i] = *(const v4f*)(src + i * 4);
}

// K tile: cvt + transpose-scatter into Ks[s][c]
__device__ __forceinline__ void store_k(_Float16 (*Ks)[KPITCH], const v4f* reg, int cL, int sh) {
#pragma unroll
  for (int i = 0; i < 8; ++i)
#pragma unroll
    for (int j = 0; j < 4; ++j)
      Ks[sh * 32 + i * 4 + j][cL] = (_Float16)reg[i][j];
}

// V tile: cvt + packed 64b stores into Vs[c][s] (memory orientation)
__device__ __forceinline__ void store_v(_Float16 (*Vs)[VPITCH], const v4f* reg, int cL, int sh) {
#pragma unroll
  for (int i = 0; i < 8; ++i) {
    union { _Float16 h[4]; uint2 u; } pk;
#pragma unroll
    for (int j = 0; j < 4; ++j) pk.h[j] = (_Float16)reg[i][j];
    *(uint2*)&Vs[cL][sh * 32 + i * 4] = pk.u;
  }
}

struct SmemBufs {
  _Float16 Qs[BR][QPITCH];          // [t][c], scale^2 folded in
  _Float16 Ks[2][BC][KPITCH];       // [s][c], double-buffered
  _Float16 Vs[2][CH][VPITCH];       // [c][s], double-buffered
  _Float16 Ps[4][16][PPITCH];       // per-wave P scratch, [t][s]
};
union SmemT {
  SmemBufs s;
  float Os[CH][OPITCH];             // output transpose tile; overlays dead Qs/Ks
};

__global__ __launch_bounds__(128)
void QKVAttention_flash_wmma(const float* __restrict__ qkv, float* __restrict__ out) {
  __shared__ SmemT sm;

  const int b    = blockIdx.y;
  const int t0   = blockIdx.x * BR;
  const int tid  = threadIdx.x;
  const int wave = tid >> 5;
  const int lane = tid & 31;
  const int ln   = lane & 15;
  const int lh   = lane >> 4;

  const float scale2 = 0.125f;      // (64^-1/4)^2 folded entirely into Q

  const float* qp = qkv + (size_t)b * C3CH * TT;
  const float* kp = qp + (size_t)CH * TT;
  const float* vp = qp + (size_t)(2 * CH) * TT;

  const int cL = tid & 63;          // channel this thread stages
  const int sh = (tid >> 6) & 1;    // which 32-key half
  const float* ksrc = kp + (size_t)cL * TT + sh * 32;
  const float* vsrc = vp + (size_t)cL * TT + sh * 32;

  // ---- prologue: stage Q tile and chunk 0 of K/V ----
  v4f kreg[8], vreg[8];
  load_chunk(ksrc, kreg);
  load_chunk(vsrc, vreg);
  {
    const int c  = tid >> 1;
    const int th = tid & 1;
    const float* src = qp + (size_t)c * TT + t0 + th * 32;
#pragma unroll
    for (int i = 0; i < 8; ++i) {
      v4f d = *(const v4f*)(src + i * 4);
#pragma unroll
      for (int j = 0; j < 4; ++j)
        sm.s.Qs[th * 32 + i * 4 + j][c] = (_Float16)(d[j] * scale2);
    }
  }
  store_k(sm.s.Ks[0], kreg, cL, sh);
  store_v(sm.s.Vs[0], vreg, cL, sh);
  __syncthreads();

  // Q fragments for this wave's 16 rows; invariant over the key loop.
  v16h aq0 = load_frag16(&sm.s.Qs[wave * 16 + ln][0], 0,  lh);
  v16h aq1 = load_frag16(&sm.s.Qs[wave * 16 + ln][0], 32, lh);

  // all-ones B fragment for matrix-pipe row sums
  v16h ones;
#pragma unroll
  for (int i = 0; i < 16; ++i) ones[i] = (_Float16)1.0f;

  v8f oacc[4] = {{}, {}, {}, {}};   // O accumulators, 4 c-tiles of 16
  float mrow[8], lrow[8], af[8];    // per-row running max / sum / rescale
#pragma unroll
  for (int r = 0; r < 8; ++r) { mrow[r] = -3.0e38f; lrow[r] = 0.f; }

  // One pipelined 64-key chunk: loads for s0+BC issued first, 18 WMMAs hide
  // the latency, converted tile stored to the alternate buffer at the end.
  auto do_chunk = [&](int pcur, int s0) {
    const bool more = (s0 + BC) < TT;
    if (more) {
      load_chunk(ksrc + s0 + BC, kreg);
      load_chunk(vsrc + s0 + BC, vreg);
    }

    // ---- scores: S = (Q^T K) for 16 rows x 64 keys, K-dim = 64 ----
    v8f sc[4];
#pragma unroll
    for (int j = 0; j < 4; ++j) {
      v8f acc = {};
      v16h bk = load_frag16(&sm.s.Ks[pcur][j * 16 + ln][0], 0, lh);
      acc = __builtin_amdgcn_wmma_f32_16x16x32_f16(false, aq0, false, bk, (short)0, acc, false, false);
      bk = load_frag16(&sm.s.Ks[pcur][j * 16 + ln][0], 32, lh);
      acc = __builtin_amdgcn_wmma_f32_16x16x32_f16(false, aq1, false, bk, (short)0, acc, false, false);
      sc[j] = acc;
    }

    // ---- online softmax; row = r + 8*lh lives across the 16 lanes of a half ----
#pragma unroll
    for (int r = 0; r < 8; ++r) {
      float cmax = rowmax16(fmaxf(fmaxf(sc[0][r], sc[1][r]), fmaxf(sc[2][r], sc[3][r])));
      float mn = fmaxf(mrow[r], cmax);
      af[r]    = __expf(mrow[r] - mn);
      mrow[r]  = mn;
#pragma unroll
      for (int j = 0; j < 4; ++j) {
        float p = __expf(sc[j][r] - mn);
        sm.s.Ps[wave][r + 8 * lh][j * 16 + ln] = (_Float16)p;
      }
#pragma unroll
      for (int ct = 0; ct < 4; ++ct) oacc[ct][r] *= af[r];
    }
    asm volatile("" ::: "memory");  // keep P stores before P reloads (same-wave DS is in-order)

    // reload P as A fragments
    v16h ap0 = load_frag16(&sm.s.Ps[wave][ln][0], 0,  lh);
    v16h ap1 = load_frag16(&sm.s.Ps[wave][ln][0], 32, lh);

    // ---- row sums on the matrix pipe: ls = P x ones, every column = row sum ----
    {
      v8f ls = {};
      ls = __builtin_amdgcn_wmma_f32_16x16x32_f16(false, ap0, false, ones, (short)0, ls, false, false);
      ls = __builtin_amdgcn_wmma_f32_16x16x32_f16(false, ap1, false, ones, (short)0, ls, false, false);
#pragma unroll
      for (int r = 0; r < 8; ++r) lrow[r] = lrow[r] * af[r] + ls[r];
    }

    // ---- O += P (16x64) x V (64x64), four 16-wide c-tiles ----
#pragma unroll
    for (int ct = 0; ct < 4; ++ct) {
      v16h bv = load_frag16(&sm.s.Vs[pcur][ct * 16 + ln][0], 0, lh);
      oacc[ct] = __builtin_amdgcn_wmma_f32_16x16x32_f16(false, ap0, false, bv, (short)0, oacc[ct], false, false);
      bv = load_frag16(&sm.s.Vs[pcur][ct * 16 + ln][0], 32, lh);
      oacc[ct] = __builtin_amdgcn_wmma_f32_16x16x32_f16(false, ap1, false, bv, (short)0, oacc[ct], false, false);
    }

    // ---- convert + store next chunk into the alternate buffer ----
    if (more) {
      store_k(sm.s.Ks[pcur ^ 1], kreg, cL, sh);
      store_v(sm.s.Vs[pcur ^ 1], vreg, cL, sh);
    }
  };

  for (int s0 = 0; s0 < TT; s0 += 2 * BC) {
    do_chunk(0, s0);
    __syncthreads();
    do_chunk(1, s0 + BC);
    __syncthreads();
  }

  // ---- epilogue: normalize, transpose through LDS, coalesced NT stores ----
#pragma unroll
  for (int r = 0; r < 8; ++r) {
    float inv = 1.0f / lrow[r];
    int   t   = wave * 16 + r + 8 * lh;
#pragma unroll
    for (int ct = 0; ct < 4; ++ct)
      sm.Os[ct * 16 + ln][t] = oacc[ct][r] * inv;
  }
  __syncthreads();
  {
    const int c  = tid >> 1;
    const int th = tid & 1;
    float* dst = out + (size_t)b * CH * TT + (size_t)c * TT + t0 + th * 32;
#pragma unroll
    for (int i = 0; i < 8; ++i) {
      v4f d = *(const v4f*)&sm.Os[c][th * 32 + i * 4];
      __builtin_nontemporal_store(d, (v4f*)(dst + i * 4));
    }
  }
}

extern "C" void kernel_launch(void* const* d_in, const int* in_sizes, int n_in,
                              void* d_out, int out_size, void* d_ws, size_t ws_size,
                              hipStream_t stream) {
  (void)in_sizes; (void)n_in; (void)out_size; (void)d_ws; (void)ws_size;
  const float* qkv = (const float*)d_in[0];
  float*       out = (float*)d_out;
  dim3 grid(TT / BR, NB, 1);   // (32, 64)
  dim3 block(128, 1, 1);       // 4 waves
  QKVAttention_flash_wmma<<<grid, block, 0, stream>>>(qkv, out);
}
